// SimpleGNN_30734785970924
// MI455X (gfx1250) — matrix operand order repaired and checked by reading, not compiled, not run on hardware
//
#include <hip/hip_runtime.h>
#include <hip/hip_bf16.h>
#include <math.h>

#define NNODES 50000
#define NEDGES 1600000
#define DIM    256
#define EPS    1e-5f
#define ROWS_PER_BLOCK 16
#define LDA (DIM + 8)          // bf16 elems; 528B row stride, 16B aligned, bank-staggered
#define LDH (DIM + 4)          // f32 elems

typedef __attribute__((ext_vector_type(16))) __bf16 v16bf;
typedef __attribute__((ext_vector_type(8)))  __bf16 v8bf;
typedef __attribute__((ext_vector_type(8)))  float  v8f;

union V16U { v16bf v; v8bf h[2]; };

// ---------------- utility kernels ----------------

__global__ void zero_f32(float* __restrict__ p, int n) {
  int i = blockIdx.x * blockDim.x + threadIdx.x;
  int stride = gridDim.x * blockDim.x;
  for (; i < n; i += stride) p[i] = 0.0f;
}

__global__ void cvt_bf16(const float* __restrict__ src, __bf16* __restrict__ dst, int n) {
  int i = blockIdx.x * blockDim.x + threadIdx.x;
  if (i < n) dst[i] = (__bf16)src[i];
}

// ---------------- edge scatter: agg[row] += x[col], deg[row] += 1 ----------------
// one thread = one float4 chunk of one edge (64 chunks/edge). x and agg are
// L2-resident (51 MB each < 192 MB L2), so the atomic traffic stays on-die.

__global__ void scatter_edges(const float* __restrict__ x,
                              const int*   __restrict__ row,
                              const int*   __restrict__ col,
                              float* __restrict__ agg,
                              float* __restrict__ deg) {
  unsigned int gid = blockIdx.x * blockDim.x + threadIdx.x;   // < E*64 = 102.4M
  int e = (int)(gid >> 6);
  int c = ((int)gid & 63) << 2;
  if (e >= NEDGES) return;
  int r = row[e];
  int s = col[e];
  const float4 v = *(const float4*)(x + (size_t)s * DIM + c);
  float* a = agg + (size_t)r * DIM + c;
  atomicAdd(a + 0, v.x);
  atomicAdd(a + 1, v.y);
  atomicAdd(a + 2, v.z);
  atomicAdd(a + 3, v.w);
  if (c == 0) atomicAdd(deg + r, 1.0f);
}

// ---------------- fused (x + agg/deg) @ W^T + b -> LayerNorm -> SiLU ----------------
// block = 256 threads = 8 waves; block computes 16 rows x 256 cols.
// wave w owns output col tiles [32w, 32w+32); K-loop of 8 wmma_f32_16x16x32_bf16 x 2 tiles.

__global__ __launch_bounds__(256)
void gnn_gemm_ln_silu(const float* __restrict__ x,
                      const float* __restrict__ agg,
                      const float* __restrict__ deg,
                      const __bf16* __restrict__ Wb,     // [DIM][DIM] row-major (n, k)
                      const float* __restrict__ bias,
                      const float* __restrict__ gamma,
                      const float* __restrict__ beta,
                      float* __restrict__ out) {
  __shared__ __attribute__((aligned(16))) __bf16 sA[ROWS_PER_BLOCK * LDA];
  __shared__ float sH[ROWS_PER_BLOCK * LDH];

  const int tid  = threadIdx.x;
  const int row0 = blockIdx.x * ROWS_PER_BLOCK;

  // Stage A = bf16(x + agg/deg) for this 16-row strip (4096 elems, 16/thread)
  for (int i = tid; i < ROWS_PER_BLOCK * DIM; i += 256) {
    int r = i >> 8;
    int c = i & 255;
    int g = row0 + r;
    float d = deg[g];
    d = d > 1.0f ? d : 1.0f;
    float v = x[(size_t)g * DIM + c] + agg[(size_t)g * DIM + c] * (1.0f / d);
    sA[r * LDA + c] = (__bf16)v;
  }
  __syncthreads();

  const int wave = tid >> 5;
  const int lane = tid & 31;
  const int half = lane >> 4;
  const int lr   = lane & 15;
  const int n0   = wave * 32;

  v8f acc0 = {};
  v8f acc1 = {};

#pragma unroll
  for (int k0 = 0; k0 < DIM; k0 += 32) {
    // A fragment (16-bit A 16x32 layout): lane half 0 -> K {0..7,16..23}, half 1 -> {8..15,24..31}
    V16U a;
    {
      const __bf16* ap = sA + lr * LDA + k0 + half * 8;
      a.h[0] = *(const v8bf*)(ap);
      a.h[1] = *(const v8bf*)(ap + 16);
    }
    // B fragments (same K order as A): lane half 0 -> K {0..7,16..23}, half 1 -> {8..15,24..31},
    // lane lr -> column n. B columns = W rows (h = a @ W^T), contiguous in memory.
    V16U b0, b1;
    {
      const __bf16* bp0 = Wb + (size_t)(n0 + lr) * DIM + k0 + half * 8;
      const __bf16* bp1 = Wb + (size_t)(n0 + 16 + lr) * DIM + k0 + half * 8;
      b0.h[0] = *(const v8bf*)(bp0);
      b0.h[1] = *(const v8bf*)(bp0 + 16);
      b1.h[0] = *(const v8bf*)(bp1);
      b1.h[1] = *(const v8bf*)(bp1 + 16);
    }
    acc0 = __builtin_amdgcn_wmma_f32_16x16x32_bf16(false, a.v, false, b0.v,
                                                   (short)0, acc0, false, false);
    acc1 = __builtin_amdgcn_wmma_f32_16x16x32_bf16(false, a.v, false, b1.v,
                                                   (short)0, acc1, false, false);
  }

  // D layout: VGPR v holds M = v + 8*half, column = n0(+16) + lr. Add bias, spill to LDS.
  {
    float bv0 = bias[n0 + lr];
    float bv1 = bias[n0 + 16 + lr];
#pragma unroll
    for (int v = 0; v < 8; ++v) {
      int m = v + 8 * half;
      sH[m * LDH + n0 + lr]      = acc0[v] + bv0;
      sH[m * LDH + n0 + 16 + lr] = acc1[v] + bv1;
    }
  }
  __syncthreads();

  // LayerNorm + SiLU: 16 threads per row, 16 cols per thread, shfl_xor reduce over 16 lanes
  {
    const int r   = tid >> 4;
    const int seg = tid & 15;
    const float* hp = sH + r * LDH + seg * 16;
    float vals[16];
    float s = 0.0f, ss = 0.0f;
#pragma unroll
    for (int i = 0; i < 16; ++i) {
      float h = hp[i];
      vals[i] = h;
      s += h;
      ss += h * h;
    }
#pragma unroll
    for (int off = 8; off >= 1; off >>= 1) {
      s  += __shfl_xor(s, off, 16);
      ss += __shfl_xor(ss, off, 16);
    }
    const float mu   = s * (1.0f / DIM);
    const float var  = ss * (1.0f / DIM) - mu * mu;
    const float rstd = rsqrtf(var + EPS);
    float* op = out + (size_t)(row0 + r) * DIM + seg * 16;
#pragma unroll
    for (int i = 0; i < 16; ++i) {
      int c = seg * 16 + i;
      float hn = (vals[i] - mu) * rstd * gamma[c] + beta[c];
      op[i] = hn / (1.0f + __expf(-hn));
    }
  }
}

// ---------------- launcher ----------------

extern "C" void kernel_launch(void* const* d_in, const int* in_sizes, int n_in,
                              void* d_out, int out_size, void* d_ws, size_t ws_size,
                              hipStream_t stream) {
  (void)in_sizes; (void)n_in; (void)out_size; (void)ws_size;

  const float* x     = (const float*)d_in[0];
  const int*   eidx  = (const int*)d_in[1];      // [2, E]
  const float* W0    = (const float*)d_in[2];
  const float* b0    = (const float*)d_in[3];
  const float* g0    = (const float*)d_in[4];
  const float* be0   = (const float*)d_in[5];
  const float* W1    = (const float*)d_in[6];
  const float* b1    = (const float*)d_in[7];
  const float* g1    = (const float*)d_in[8];
  const float* be1   = (const float*)d_in[9];
  float* out = (float*)d_out;

  const int* row = eidx;
  const int* col = eidx + NEDGES;

  // workspace layout: agg [N*256] f32 | deg [N] f32 | Wb0 [256*256] bf16 | Wb1 [256*256] bf16
  float*  agg = (float*)d_ws;
  float*  deg = agg + (size_t)NNODES * DIM;
  __bf16* Wb0 = (__bf16*)(deg + NNODES);
  __bf16* Wb1 = Wb0 + DIM * DIM;

  const int zero_n = NNODES * DIM + NNODES;     // agg + deg contiguous

  // weights -> bf16 (once per launch; deterministic)
  cvt_bf16<<<(DIM * DIM + 255) / 256, 256, 0, stream>>>(W0, Wb0, DIM * DIM);
  cvt_bf16<<<(DIM * DIM + 255) / 256, 256, 0, stream>>>(W1, Wb1, DIM * DIM);

  const dim3 scatter_grid((unsigned)(((size_t)NEDGES * (DIM / 4) + 255) / 256));
  const dim3 gemm_grid(NNODES / ROWS_PER_BLOCK);   // 3125, exact

  // ---- layer 1: x -> out ----
  zero_f32<<<16384, 256, 0, stream>>>(agg, zero_n);
  scatter_edges<<<scatter_grid, 256, 0, stream>>>(x, row, col, agg, deg);
  gnn_gemm_ln_silu<<<gemm_grid, 256, 0, stream>>>(x, agg, deg, Wb0, b0, g0, be0, out);

  // ---- layer 2: out -> out (blocks touch only their own 16-row strip) ----
  zero_f32<<<16384, 256, 0, stream>>>(agg, zero_n);
  scatter_edges<<<scatter_grid, 256, 0, stream>>>(out, row, col, agg, deg);
  gnn_gemm_ln_silu<<<gemm_grid, 256, 0, stream>>>(out, agg, deg, Wb1, b1, g1, be1, out);
}
